// CoattentiveAggregation_90177133346926
// MI455X (gfx1250) — matrix operand order repaired and checked by reading, not compiled, not run on hardware
//
#include <hip/hip_runtime.h>

// CDNA5 / gfx1250, wave32. One wave per destination node.
typedef _Float16 v16h __attribute__((ext_vector_type(16)));
typedef _Float16 v8h  __attribute__((ext_vector_type(8)));
typedef float    v8f  __attribute__((ext_vector_type(8)));

#define MN   32      // neighbors (M == K == 32)
#define FD   128     // feature dim
#define LDH  136     // padded LDS row stride in halves (272B: breaks 64-bank stride, keeps 16B align)
#define LDL  33      // padded LDS row stride for L (floats)

__global__ __launch_bounds__(32)
void coatt_agg_kernel(const float* __restrict__ feat,
                      const int*   __restrict__ idx_sim,
                      const int*   __restrict__ idx_cor,
                      float*       __restrict__ out)
{
    __shared__ __align__(16) _Float16 Dh[MN][LDH];   // gathered D rows, f16
    __shared__ __align__(16) _Float16 Qh[MN][LDH];   // gathered Q rows, f16
    __shared__ __align__(16) float    Lm[MN][LDL];   // L = D * Q^T
    __shared__ float colmax[MN], Zc[MN], rowmax[MN], Zr[MN], wv[MN], uv[MN];
    __shared__ float S[3 * FD];                      // [colsum(Q) | w@D | u@Q]

    const int n    = blockIdx.x;
    const int lane = threadIdx.x & 31;
    const int hi   = lane >> 4;      // half-wave select
    const int l15  = lane & 15;

    // ---- Phase 1: gather one D row and one Q row per lane, f32 -> f16 into LDS
    {
        const int rs = idx_sim[n * MN + lane];
        const int rc = idx_cor[n * MN + lane];
        const float4* __restrict__ ds = (const float4*)(feat + (size_t)rs * FD);
        const float4* __restrict__ qs = (const float4*)(feat + (size_t)rc * FD);
        #pragma unroll
        for (int f4 = 0; f4 < FD / 4; ++f4) {
            float4 d = ds[f4];
            float4 q = qs[f4];
            Dh[lane][f4*4+0] = (_Float16)d.x; Dh[lane][f4*4+1] = (_Float16)d.y;
            Dh[lane][f4*4+2] = (_Float16)d.z; Dh[lane][f4*4+3] = (_Float16)d.w;
            Qh[lane][f4*4+0] = (_Float16)q.x; Qh[lane][f4*4+1] = (_Float16)q.y;
            Qh[lane][f4*4+2] = (_Float16)q.z; Qh[lane][f4*4+3] = (_Float16)q.w;
        }
    }
    __syncthreads();   // single-wave WG: lowers to waitcnt-only, no barrier resource

    // ---- Phase 2: L = D * Q^T via v_wmma_f32_16x16x32_f16 (2x2 tiles, 4 k-chunks)
    union ABFrag { v16h v; v8h h[2]; };
    #pragma unroll
    for (int mt = 0; mt < 2; ++mt) {
        #pragma unroll
        for (int kt = 0; kt < 2; ++kt) {
            v8f acc = {0.f, 0.f, 0.f, 0.f, 0.f, 0.f, 0.f, 0.f};
            #pragma unroll
            for (int f0 = 0; f0 < FD; f0 += 32) {
                // A fragment: D rows mt*16+l15; ISA layout K=(i&7) + 16*(i>=8) + 8*hi
                const _Float16* arow = &Dh[mt*16 + l15][0];
                ABFrag a;
                a.h[0] = *(const v8h*)(arow + f0 + 8*hi);
                a.h[1] = *(const v8h*)(arow + f0 + 16 + 8*hi);
                // B fragment: column k = Q row kt*16+l15; ISA layout K=(i&7) + 16*(i>=8) + 8*hi
                const _Float16* brow = &Qh[kt*16 + l15][0];
                ABFrag b;
                b.h[0] = *(const v8h*)(brow + f0 + 8*hi);
                b.h[1] = *(const v8h*)(brow + f0 + 16 + 8*hi);
                acc = __builtin_amdgcn_wmma_f32_16x16x32_f16(
                        /*neg_a=*/false, a.v, /*neg_b=*/false, b.v,
                        /*c_mod=*/(short)0, acc, /*reuse_a=*/false, /*reuse_b=*/false);
            }
            // C/D layout: VGPR r -> row M = r + 8*hi, col N = l15
            #pragma unroll
            for (int r = 0; r < 8; ++r)
                Lm[mt*16 + 8*hi + r][kt*16 + l15] = acc[r];
        }
    }
    __syncthreads();

    // ---- Phase 3: softmax statistics
    {   // lane = column i (column softmax partition Zc) and row m (row softmax Zr)
        float cmax = -__builtin_inff();
        for (int j = 0; j < MN; ++j) cmax = fmaxf(cmax, Lm[j][lane]);
        float zc = 0.f;
        for (int j = 0; j < MN; ++j) zc += __expf(Lm[j][lane] - cmax);
        colmax[lane] = cmax;  Zc[lane] = zc;

        float rmax = -__builtin_inff();
        for (int k = 0; k < MN; ++k) rmax = fmaxf(rmax, Lm[lane][k]);
        float zr = 0.f;
        for (int k = 0; k < MN; ++k) zr += __expf(Lm[lane][k] - rmax);
        rowmax[lane] = rmax;  Zr[lane] = zr;
    }
    __syncthreads();
    {   // w[j] = sum_i exp(L[j,i]-colmax[i]) / Zc[i]   (column-softmax column sums)
        float w = 0.f;
        for (int i = 0; i < MN; ++i)
            w += __expf(Lm[lane][i] - colmax[i]) / Zc[i];
        wv[lane] = w;
    }
    __syncthreads();
    {   // u[r] = sum_j w[j] * exp(L[j,r]-rowmax[j]) / Zr[j]   (u = w @ AC)
        float u = 0.f;
        for (int j = 0; j < MN; ++j)
            u += wv[j] * __expf(Lm[j][lane] - rowmax[j]) / Zr[j];
        uv[lane] = u;
    }
    __syncthreads();

    // ---- Phase 4: S = [colsum(Q) | w@D | u@Q] ; each lane owns 4 feature columns
    {
        const int c0 = 4 * lane;
        float sq[4] = {0.f,0.f,0.f,0.f}, t1[4] = {0.f,0.f,0.f,0.f}, t2[4] = {0.f,0.f,0.f,0.f};
        for (int j = 0; j < MN; ++j) {
            const float w = wv[j], u = uv[j];
            #pragma unroll
            for (int t = 0; t < 4; ++t) {
                const float d = (float)Dh[j][c0 + t];
                const float q = (float)Qh[j][c0 + t];
                sq[t] += q;
                t1[t] += w * d;
                t2[t] += u * q;
            }
        }
        #pragma unroll
        for (int t = 0; t < 4; ++t) {
            S[c0 + t]            = sq[t];
            S[FD + c0 + t]       = t1[t];
            S[2 * FD + c0 + t]   = t2[t];
        }
    }
    __syncthreads();

    // ---- Phase 5: pooled/mean combine + self feature
    {
        const int c0 = 4 * lane;
        #pragma unroll
        for (int t = 0; t < 4; ++t) {
            const int c = c0 + t;
            const float h = (S[3*c] + S[3*c + 1] + S[3*c + 2]) * (1.0f / 96.0f);
            out[(size_t)n * FD + c] = feat[(size_t)n * FD + c] + h;
        }
    }
}

extern "C" void kernel_launch(void* const* d_in, const int* in_sizes, int n_in,
                              void* d_out, int out_size, void* d_ws, size_t ws_size,
                              hipStream_t stream) {
    (void)n_in; (void)d_ws; (void)ws_size; (void)out_size;
    const float* feat    = (const float*)d_in[0];
    const int*   idx_sim = (const int*)d_in[1];
    const int*   idx_cor = (const int*)d_in[2];
    float*       out     = (float*)d_out;
    const int n_dst = in_sizes[1] / MN;   // 10000
    coatt_agg_kernel<<<dim3(n_dst), dim3(32), 0, stream>>>(feat, idx_sim, idx_cor, out);
}
